// ResRNN_52192442581864
// MI455X (gfx1250) — matrix operand order repaired and checked by reading, not compiled, not run on hardware
//
#include <hip/hip_runtime.h>
#include <hip/hip_bf16.h>
#include <math.h>

#define BSZ 64
#define TSZ 512
#define ISZ 1024
#define HSZ 4096
#define KCAT 2048   // ISZ + OSIZE

typedef __attribute__((ext_vector_type(16))) __bf16 v16bf;
typedef __attribute__((ext_vector_type(8)))  float  v8f;

__device__ __forceinline__ unsigned short f2bf(float f) {
    union { float f; unsigned u; } v; v.f = f;
    unsigned u = v.u;
    u += 0x7FFFu + ((u >> 16) & 1u);   // round-to-nearest-even
    return (unsigned short)(u >> 16);
}

__device__ __forceinline__ unsigned pk2(float lo, float hi) {
    return (unsigned)f2bf(lo) | ((unsigned)f2bf(hi) << 16);
}

#define WMMA_BF16(A, Bm, C) \
    __builtin_amdgcn_wmma_f32_16x16x32_bf16(false, (A), false, (Bm), (short)0, (C), false, false)

// ---------------------------------------------------------------------------
// Pack fp32 weight [K x N] row-major into WMMA B-operand tiles:
// tile (kt, nt) of 32x16; lane = (n%16) + 16*((k/8)%2); pos = 8*(k%32 >= 16) + k%8.
// Each lane's 16 bf16 values are contiguous (32B) -> 2x global_load_b128.
// ---------------------------------------------------------------------------
__global__ void pack_b_kernel(const float* __restrict__ W, unsigned short* __restrict__ out,
                              int K, int N) {
    size_t idx = (size_t)blockIdx.x * blockDim.x + threadIdx.x;
    if (idx >= (size_t)K * N) return;
    int n = (int)(idx % N);
    int k = (int)(idx / N);
    int kt = k >> 5, kr = k & 31;
    int nt = n >> 4, nl = n & 15;
    int lane = nl + (((kr >> 3) & 1) ? 16 : 0);
    int pos  = (kr >> 4) * 8 + (kr & 7);
    out[((size_t)kt * (N >> 4) + nt) * 512 + lane * 16 + pos] = f2bf(W[idx]);
}

// ---------------------------------------------------------------------------
// A-operand bf16 layout (16x32 tile): lane<16 holds row m%16, K {0..7,16..23};
// lane>=16 holds the same row, K {8..15,24..31}; 16 contiguous bf16 per lane.
// ---------------------------------------------------------------------------
__device__ __forceinline__ void pack_a_elem(unsigned short* __restrict__ apk,
                                            int m, int k, int mtiles, unsigned short v) {
    int kt = k >> 5, kr = k & 31;
    int lo = kr & 15, hi = kr >> 4;
    int lane = (m & 15) + ((lo < 8) ? 0 : 16);
    int pos  = hi * 8 + (lo & 7);
    apk[((size_t)(kt * mtiles + (m >> 4))) * 512 + lane * 16 + pos] = v;
}

__global__ void init_hx_kernel(const float* __restrict__ init_hx,
                               float* __restrict__ hxf, unsigned short* __restrict__ hxpk) {
    int idx = blockIdx.x * 256 + threadIdx.x;       // 0 .. 64*1024-1
    int m = idx >> 10, n = idx & 1023;
    float v = init_hx[n];
    hxf[idx] = v;
    pack_a_elem(hxpk, m, ISZ + n, 4, f2bf(v));
}

// ---------------------------------------------------------------------------
// GEMM1: h = cat(x_t, hx) @ W1 + b1.   M=64, N=4096, K=2048.
// Block = 8 waves, one M-tile; waves cover 8 N-strips of 4 N-tiles.
// A operand staged in LDS per 32-kt chunk (32 KB):
//   chunk 0: packed on the fly from fp32 x[:, t, :]
//   chunk 1: copied from the bf16 hx buffer written by the previous GEMM2.
// B operand streamed register-direct from L2, software pipelined.
// grid = (8, 4): blockIdx.y = M-tile.
// ---------------------------------------------------------------------------
__global__ void gemm1_kernel(const float* __restrict__ x, int t,
                             const unsigned short* __restrict__ hxpk,
                             const unsigned short* __restrict__ w1pk,
                             const float* __restrict__ b1,
                             float* __restrict__ hraw) {
    __shared__ unsigned short lA[32 * 512];   // 32 KB: 32 A tiles of 1 KB

    const int tid  = threadIdx.x;
    const int lane = tid & 31;
    const int wave = tid >> 5;
    const int mt   = blockIdx.y;
    const int nt0  = (blockIdx.x * 8 + wave) * 4;
    const int l    = lane & 15;
    const int hi   = lane >> 4;

    v8f c0 = {}, c1 = {}, c2 = {}, c3 = {};

    for (int chunk = 0; chunk < 2; ++chunk) {
        // ---- cooperative stage of 32 A tiles into LDS ----
        if (chunk == 0) {
            // pack rows [mt*16, mt*16+16) of x[:, t, :] (fp32 -> bf16 A layout)
            #pragma unroll
            for (int i = 0; i < 8; ++i) {
                int rI   = i * 256 + tid;          // 2048 runs of 8 elements
                int lr   = rI >> 7;                // local row 0..15
                int kblk = (rI & 127) * 8;         // k base, multiple of 8
                const float* src = x + ((size_t)(mt * 16 + lr) * TSZ + t) * ISZ + kblk;
                float4 f0 = *(const float4*)(src);
                float4 f1 = *(const float4*)(src + 4);
                uint4 pk;
                pk.x = pk2(f0.x, f0.y); pk.y = pk2(f0.z, f0.w);
                pk.z = pk2(f1.x, f1.y); pk.w = pk2(f1.z, f1.w);
                int kt   = kblk >> 5, kr = kblk & 31;
                int lo   = kr & 15,   hb = kr >> 4;
                int ln2  = lr + ((lo < 8) ? 0 : 16);
                int pos  = hb * 8;
                *(uint4*)&lA[kt * 512 + ln2 * 16 + pos] = pk;
            }
        } else {
            // copy 32 KB of packed hx state (global tiles kt 32..63, this mt)
            #pragma unroll
            for (int i = 0; i < 8; ++i) {
                int c      = i * 256 + tid;        // 2048 chunks of 16 B
                int tile   = c >> 6;               // 0..31
                int within = c & 63;               // 16B chunk within tile
                const uint4* src = (const uint4*)(hxpk +
                    ((size_t)((32 + tile) * 4 + mt)) * 512 + within * 8);
                *(uint4*)&lA[tile * 512 + within * 8] = *src;
            }
        }
        __syncthreads();

        // ---- 32-kt compute with B software pipeline ----
        const unsigned short* bp = w1pk + ((size_t)chunk * 32 * (HSZ / 16) + nt0) * 512 + lane * 16;
        v16bf bm0 = *(const v16bf*)(bp);
        v16bf bm1 = *(const v16bf*)(bp + 512);
        v16bf bm2 = *(const v16bf*)(bp + 1024);
        v16bf bm3 = *(const v16bf*)(bp + 1536);

        for (int kt2 = 0; kt2 < 31; ++kt2) {
            bp += (HSZ / 16) * 512;
            v16bf n0 = *(const v16bf*)(bp);
            v16bf n1 = *(const v16bf*)(bp + 512);
            v16bf n2 = *(const v16bf*)(bp + 1024);
            v16bf n3 = *(const v16bf*)(bp + 1536);
            v16bf a  = *(const v16bf*)&lA[kt2 * 512 + lane * 16];
            c0 = WMMA_BF16(a, bm0, c0);
            c1 = WMMA_BF16(a, bm1, c1);
            c2 = WMMA_BF16(a, bm2, c2);
            c3 = WMMA_BF16(a, bm3, c3);
            bm0 = n0; bm1 = n1; bm2 = n2; bm3 = n3;
        }
        {
            v16bf a = *(const v16bf*)&lA[31 * 512 + lane * 16];
            c0 = WMMA_BF16(a, bm0, c0);
            c1 = WMMA_BF16(a, bm1, c1);
            c2 = WMMA_BF16(a, bm2, c2);
            c3 = WMMA_BF16(a, bm3, c3);
        }
        __syncthreads();
    }

    // C/D layout: VGPR v, lanes 0-15 -> M=v, lanes 16-31 -> M=v+8; col = lane%16.
    const int rowb = mt * 16 + hi * 8;
    v8f cs[4] = { c0, c1, c2, c3 };
    #pragma unroll
    for (int j = 0; j < 4; ++j) {
        int col = (nt0 + j) * 16 + l;
        float bias = b1[col];
        #pragma unroll
        for (int v = 0; v < 8; ++v)
            hraw[(size_t)(rowb + v) * HSZ + col] = cs[j][v] + bias;
    }
}

// ---------------------------------------------------------------------------
// Per-row LayerNorm + exact GELU, output packed as WMMA A-operand bf16 tiles
// for GEMM2 (K = 4096, 4 M-tiles). One block (256 thr) per batch row.
// ---------------------------------------------------------------------------
__global__ void ln_gelu_pack_kernel(const float* __restrict__ hraw,
                                    const float* __restrict__ ln_g,
                                    const float* __restrict__ ln_b,
                                    unsigned short* __restrict__ hpk) {
    const int m = blockIdx.x;
    const int tid = threadIdx.x;
    const float* row = hraw + (size_t)m * HSZ;

    float s = 0.f, s2 = 0.f;
    for (int i = tid; i < HSZ; i += 256) { float v = row[i]; s += v; s2 += v * v; }
    __shared__ float sh[256], sh2[256];
    sh[tid] = s; sh2[tid] = s2;
    __syncthreads();
    for (int o = 128; o > 0; o >>= 1) {
        if (tid < o) { sh[tid] += sh[tid + o]; sh2[tid] += sh2[tid + o]; }
        __syncthreads();
    }
    const float mu   = sh[0] * (1.0f / HSZ);
    const float var  = sh2[0] * (1.0f / HSZ) - mu * mu;
    const float rinv = rsqrtf(var + 1e-6f);

    for (int i = tid; i < HSZ; i += 256) {
        float v = (row[i] - mu) * rinv * ln_g[i] + ln_b[i];
        float g = 0.5f * v * (1.0f + erff(v * 0.70710678118654752f));  // exact GELU
        pack_a_elem(hpk, m, i, 4, f2bf(g));
    }
}

// ---------------------------------------------------------------------------
// GEMM2: hx_new = h @ W2 + b2 + hx.   M=64, N=1024, K=4096.
// A staged in LDS per 32-kt chunk (4 chunks); B pipelined from L2.
// Writes out[:, t, :], updates fp32 hx in place, and re-packs hx into the
// bf16 A-layout for the next step's GEMM1.
// grid = (2, 4): blockIdx.y = M-tile, 16 strips of 4 N-tiles.
// ---------------------------------------------------------------------------
__global__ void gemm2_kernel(const unsigned short* __restrict__ hpk,
                             const unsigned short* __restrict__ w2pk,
                             const float* __restrict__ b2,
                             float* __restrict__ hxf,
                             unsigned short* __restrict__ hxpk,
                             float* __restrict__ out, int t) {
    __shared__ unsigned short lA[32 * 512];   // 32 KB

    const int tid  = threadIdx.x;
    const int lane = tid & 31;
    const int wave = tid >> 5;
    const int mt   = blockIdx.y;
    const int nt0  = (blockIdx.x * 8 + wave) * 4;
    const int l    = lane & 15;
    const int hi   = lane >> 4;

    v8f c0 = {}, c1 = {}, c2 = {}, c3 = {};

    for (int chunk = 0; chunk < 4; ++chunk) {
        // ---- cooperative copy of 32 A tiles (this mt) into LDS ----
        #pragma unroll
        for (int i = 0; i < 8; ++i) {
            int c      = i * 256 + tid;            // 2048 chunks of 16 B
            int tile   = c >> 6;
            int within = c & 63;
            const uint4* src = (const uint4*)(hpk +
                ((size_t)((chunk * 32 + tile) * 4 + mt)) * 512 + within * 8);
            *(uint4*)&lA[tile * 512 + within * 8] = *src;
        }
        __syncthreads();

        const unsigned short* bp = w2pk + ((size_t)chunk * 32 * (ISZ / 16) + nt0) * 512 + lane * 16;
        v16bf bm0 = *(const v16bf*)(bp);
        v16bf bm1 = *(const v16bf*)(bp + 512);
        v16bf bm2 = *(const v16bf*)(bp + 1024);
        v16bf bm3 = *(const v16bf*)(bp + 1536);

        for (int kt2 = 0; kt2 < 31; ++kt2) {
            bp += (ISZ / 16) * 512;
            v16bf n0 = *(const v16bf*)(bp);
            v16bf n1 = *(const v16bf*)(bp + 512);
            v16bf n2 = *(const v16bf*)(bp + 1024);
            v16bf n3 = *(const v16bf*)(bp + 1536);
            v16bf a  = *(const v16bf*)&lA[kt2 * 512 + lane * 16];
            c0 = WMMA_BF16(a, bm0, c0);
            c1 = WMMA_BF16(a, bm1, c1);
            c2 = WMMA_BF16(a, bm2, c2);
            c3 = WMMA_BF16(a, bm3, c3);
            bm0 = n0; bm1 = n1; bm2 = n2; bm3 = n3;
        }
        {
            v16bf a = *(const v16bf*)&lA[31 * 512 + lane * 16];
            c0 = WMMA_BF16(a, bm0, c0);
            c1 = WMMA_BF16(a, bm1, c1);
            c2 = WMMA_BF16(a, bm2, c2);
            c3 = WMMA_BF16(a, bm3, c3);
        }
        __syncthreads();
    }

    const int rowb = mt * 16 + hi * 8;
    v8f cs[4] = { c0, c1, c2, c3 };
    #pragma unroll
    for (int j = 0; j < 4; ++j) {
        int col = (nt0 + j) * 16 + l;
        float bias = b2[col];
        #pragma unroll
        for (int v = 0; v < 8; ++v) {
            int row = rowb + v;
            size_t hix = (size_t)row * ISZ + col;
            float val = cs[j][v] + bias + hxf[hix];
            out[((size_t)row * TSZ + t) * ISZ + col] = val;
            hxf[hix] = val;
            pack_a_elem(hxpk, row, ISZ + col, 4, f2bf(val));
        }
    }
}

// ---------------------------------------------------------------------------
extern "C" void kernel_launch(void* const* d_in, const int* in_sizes, int n_in,
                              void* d_out, int out_size, void* d_ws, size_t ws_size,
                              hipStream_t stream) {
    const float* x       = (const float*)d_in[0];
    const float* W1      = (const float*)d_in[1];
    const float* b1      = (const float*)d_in[2];
    const float* ln_g    = (const float*)d_in[3];
    const float* ln_b    = (const float*)d_in[4];
    const float* W2      = (const float*)d_in[5];
    const float* b2      = (const float*)d_in[6];
    const float* init_hx = (const float*)d_in[7];
    float* out = (float*)d_out;

    char* w = (char*)d_ws;
    unsigned short* w1pk = (unsigned short*)(w);                                 // 16 MB
    unsigned short* w2pk = (unsigned short*)(w + (16u << 20));                   //  8 MB
    float*          hxf  = (float*)(w + (24u << 20));                            // 256 KB
    unsigned short* hxpk = (unsigned short*)(w + (24u << 20) + (256u << 10));    // 256 KB (hx region, A layout)
    float*          hraw = (float*)(w + (24u << 20) + (512u << 10));             //   1 MB
    unsigned short* hpk  = (unsigned short*)(w + (24u << 20) + (512u << 10) + (1u << 20)); // 512 KB

    // Pack weights to WMMA B-operand bf16 layout (deterministic, every call).
    pack_b_kernel<<<(KCAT * HSZ + 255) / 256, 256, 0, stream>>>(W1, w1pk, KCAT, HSZ);
    pack_b_kernel<<<(HSZ * ISZ + 255) / 256, 256, 0, stream>>>(W2, w2pk, HSZ, ISZ);

    // hx state init (fp32 + packed bf16).
    init_hx_kernel<<<(BSZ * ISZ) / 256, 256, 0, stream>>>(init_hx, hxf, hxpk);

    for (int t = 0; t < TSZ; ++t) {
        gemm1_kernel<<<dim3(8, 4), 256, 0, stream>>>(x, t, hxpk, w1pk, b1, hraw);
        ln_gelu_pack_kernel<<<dim3(BSZ), 256, 0, stream>>>(hraw, ln_g, ln_b, hpk);
        gemm2_kernel<<<dim3(2, 4), 256, 0, stream>>>(hpk, w2pk, b2, hxf, hxpk, out, t);
    }
}
